// MultiheadSelfAttention_CRAT_15436112462395
// MI455X (gfx1250) — compile-verified, matrix operand order checked by reading, not run on hardware
//
#include <hip/hip_runtime.h>
#include <hip/hip_bf16.h>

typedef __attribute__((ext_vector_type(16))) _Float16 v16h;
typedef __attribute__((ext_vector_type(8)))  _Float16 v8h;
typedef __attribute__((ext_vector_type(8)))  float    v8f;

#define NB    1024
#define LMAX  64
#define DM    128
#define NH    4
#define HDIM  32

// ---- LDS layout (dynamic, 106496 bytes total) ----
// Xh / Ch : [64][136] f16  @ 0       (17408 B)   input X, later ctx
// QKh/outb: [64][264] f16  @ 17408   (33792 B)   Q(0..127) K(128..255); later f32 [64][132] out staging
// Vt      : [128][72] f16  @ 51200   (18432 B)   V transposed: [hd][key]
// Sh      : [4][64][72] f16 @ 69632  (36864 B)   attn weights per head
#define SM_QK   17408
#define SM_VT   51200
#define SM_SH   69632
#define SM_TOT  106496

__device__ __forceinline__ v16h ldfrag(const _Float16* p) {
    // 16-bit WMMA operand fragment: lo = 8 halves at p, hi = 8 halves at p+16
    union { v16h v; struct { v8h lo, hi; } s; } u;
    u.s.lo = *(const v8h*)(p);
    u.s.hi = *(const v8h*)(p + 16);
    return u.v;
}

__device__ __forceinline__ v8f wmma16(v16h a, v16h b, v8f c) {
    return __builtin_amdgcn_wmma_f32_16x16x32_f16(false, a, false, b, (short)0, c, false, false);
}

// ---------------- kernel 0: exclusive prefix sum of counts ----------------
__global__ void scan_offsets(const int* __restrict__ counts, int* __restrict__ offsets) {
    __shared__ int buf[NB];
    int t = threadIdx.x;
    int c = counts[t];
    buf[t] = c;
    __syncthreads();
    for (int d = 1; d < NB; d <<= 1) {
        int v = (t >= d) ? buf[t - d] : 0;
        __syncthreads();
        buf[t] += v;
        __syncthreads();
    }
    offsets[t] = buf[t] - c;
}

// ---------------- kernel 1: fp32 -> f16 weight conversion ----------------
__global__ void convert_weights(const float* __restrict__ Wqkv, const float* __restrict__ Wout,
                                _Float16* __restrict__ Wqkvh, _Float16* __restrict__ Wouth) {
    const int n1 = 3 * DM * DM;   // 49152
    const int n2 = DM * DM;       // 16384
    int i = blockIdx.x * blockDim.x + threadIdx.x;
    if (i < n1)            Wqkvh[i]       = (_Float16)Wqkv[i];
    else if (i < n1 + n2)  Wouth[i - n1]  = (_Float16)Wout[i - n1];
}

// ---------------- kernel 2: fused ragged MHSA, one block per scene ----------------
__global__ __launch_bounds__(256) void mhsa_kernel(
    const float* __restrict__ att_in,      // [TOTAL][128]
    const float* __restrict__ in_bias,     // [384]
    const float* __restrict__ out_bias,    // [128]
    const int*   __restrict__ counts,      // [B]
    const int*   __restrict__ offsets,     // [B]
    const _Float16* __restrict__ Wqkvh,    // [384][128] f16
    const _Float16* __restrict__ Wouth,    // [128][128] f16
    float* __restrict__ out)               // [B][64][128]
{
    extern __shared__ __align__(16) char smem[];
    _Float16* Xh   = (_Float16*)(smem);            // [64][136]
    _Float16* QKh  = (_Float16*)(smem + SM_QK);    // [64][264]
    float*    outb = (float*)   (smem + SM_QK);    // [64][132]  (aliases QKh)
    _Float16* Vt   = (_Float16*)(smem + SM_VT);    // [128][72]
    _Float16* Sh   = (_Float16*)(smem + SM_SH);    // [4][64][72]
    _Float16* Ch   = Xh;                           // ctx aliases X

    const int b     = blockIdx.x;
    const int t     = threadIdx.x;
    const int wave  = t >> 5;
    const int lane  = t & 31;
    const int laneN = lane & 15;
    const int laneH = lane >> 4;
    const int n     = counts[b];
    const int off   = offsets[b];

    // ---- Phase 1: gather + pad X into LDS as f16 (coalesced) ----
    #pragma unroll
    for (int i = 0; i < 32; ++i) {
        int e = i * 256 + t;
        int l = e >> 7, c = e & 127;
        float v = (l < n) ? att_in[(size_t)(off + l) * DM + c] : 0.0f;
        Xh[l * 136 + c] = (_Float16)v;
    }
    __syncthreads();

    // ---- Phase 2: QKV projection  qkv[l][j] = sum_d X[l][d]*W[j][d] + bias[j] ----
    // 24 N-tiles (384 cols), 4 M-tiles; wave w handles N-tiles 3w..3w+2.
    for (int nt = wave * 3; nt < wave * 3 + 3; ++nt) {
        const int j = nt * 16 + laneN;             // output column (B-operand lane)
        const float bj = in_bias[j];
        #pragma unroll
        for (int mt = 0; mt < 4; ++mt) {
            v8f acc;
            #pragma unroll
            for (int r = 0; r < 8; ++r) acc[r] = bj;
            #pragma unroll
            for (int kb = 0; kb < 4; ++kb) {
                v16h a  = ldfrag(Xh + (mt * 16 + laneN) * 136 + kb * 32 + laneH * 8);
                v16h bf = ldfrag(Wqkvh + j * DM + kb * 32 + laneH * 8);
                acc = wmma16(a, bf, acc);
            }
            if (nt < 16) {                         // Q or K -> row-major [l][j]
                #pragma unroll
                for (int r = 0; r < 8; ++r)
                    QKh[(mt * 16 + laneH * 8 + r) * 264 + j] = (_Float16)acc[r];
            } else {                               // V -> transposed [hd][key], 16B packed store
                v8h pk;
                #pragma unroll
                for (int r = 0; r < 8; ++r) pk[r] = (_Float16)acc[r];
                *(v8h*)(Vt + (j - 256) * 72 + mt * 16 + laneH * 8) = pk;
            }
        }
    }
    __syncthreads();

    // ---- Phase 3: scores = (Q*scale)·K^T, mask, softmax -> Sh ----
    {
        const int h = wave >> 1, half = wave & 1;
        const float scale = 0.17677669529663687f;  // 1/sqrt(32)
        #pragma unroll
        for (int mti = 0; mti < 2; ++mti) {
            const int mt = half * 2 + mti;         // this wave's query row-tile
            v16h a = ldfrag(QKh + (mt * 16 + laneN) * 264 + h * HDIM + laneH * 8);
            v8f s[4];
            #pragma unroll
            for (int nt = 0; nt < 4; ++nt) {       // key tiles, K=32 -> single WMMA each
                v16h bf = ldfrag(QKh + (nt * 16 + laneN) * 264 + DM + h * HDIM + laneH * 8);
                v8f z = {};
                s[nt] = wmma16(a, bf, z);
            }
            // scale + key-padding mask (per-lane select, EXEC stays uniform)
            #pragma unroll
            for (int nt = 0; nt < 4; ++nt) {
                bool valid = (nt * 16 + laneN) < n;
                #pragma unroll
                for (int r = 0; r < 8; ++r)
                    s[nt][r] = valid ? s[nt][r] * scale : -1e30f;
            }
            // row max: across 4 tiles then across each 16-lane half (rows differ per half)
            v8f rmx;
            #pragma unroll
            for (int r = 0; r < 8; ++r) {
                float v = fmaxf(fmaxf(s[0][r], s[1][r]), fmaxf(s[2][r], s[3][r]));
                v = fmaxf(v, __shfl_xor(v, 1, 32));
                v = fmaxf(v, __shfl_xor(v, 2, 32));
                v = fmaxf(v, __shfl_xor(v, 4, 32));
                v = fmaxf(v, __shfl_xor(v, 8, 32));
                rmx[r] = v;
            }
            // exp + row sum
            v8f rsum;
            #pragma unroll
            for (int r = 0; r < 8; ++r) {
                float acc = 0.0f;
                #pragma unroll
                for (int nt = 0; nt < 4; ++nt) {
                    s[nt][r] = __expf(s[nt][r] - rmx[r]);
                    acc += s[nt][r];
                }
                acc += __shfl_xor(acc, 1, 32);
                acc += __shfl_xor(acc, 2, 32);
                acc += __shfl_xor(acc, 4, 32);
                acc += __shfl_xor(acc, 8, 32);
                rsum[r] = 1.0f / acc;
            }
            // normalized attn -> f16 LDS [h][row][key]
            #pragma unroll
            for (int nt = 0; nt < 4; ++nt) {
                int key = nt * 16 + laneN;
                #pragma unroll
                for (int r = 0; r < 8; ++r)
                    Sh[(h * 64 + mt * 16 + laneH * 8 + r) * 72 + key] =
                        (_Float16)(s[nt][r] * rsum[r]);
            }
        }
    }
    __syncthreads();

    // ---- Phase 4: ctx = attn · V  (K=64 keys, 2 WMMAs per tile) ----
    {
        const int h = wave >> 1, half = wave & 1;
        #pragma unroll
        for (int mti = 0; mti < 2; ++mti) {
            const int mt = half * 2 + mti;
            #pragma unroll
            for (int nt = 0; nt < 2; ++nt) {       // 32 hd cols = 2 tiles
                v8f acc = {};
                #pragma unroll
                for (int kb = 0; kb < 2; ++kb) {
                    v16h a  = ldfrag(Sh + (h * 64 + mt * 16 + laneN) * 72 + kb * 32 + laneH * 8);
                    v16h bf = ldfrag(Vt + (h * HDIM + nt * 16 + laneN) * 72 + kb * 32 + laneH * 8);
                    acc = wmma16(a, bf, acc);
                }
                int col = h * HDIM + nt * 16 + laneN;
                #pragma unroll
                for (int r = 0; r < 8; ++r)
                    Ch[(mt * 16 + laneH * 8 + r) * 136 + col] = (_Float16)acc[r];
            }
        }
    }
    __syncthreads();

    // ---- Phase 5: out projection  out = ctx · Wout^T + bias ----
    {
        const int nt = wave;                       // 8 column tiles, one per wave
        const int j = nt * 16 + laneN;
        const float bj = out_bias[j];
        #pragma unroll
        for (int mt = 0; mt < 4; ++mt) {
            v8f acc;
            #pragma unroll
            for (int r = 0; r < 8; ++r) acc[r] = bj;
            #pragma unroll
            for (int kb = 0; kb < 4; ++kb) {
                v16h a  = ldfrag(Ch + (mt * 16 + laneN) * 136 + kb * 32 + laneH * 8);
                v16h bf = ldfrag(Wouth + j * DM + kb * 32 + laneH * 8);
                acc = wmma16(a, bf, acc);
            }
            #pragma unroll
            for (int r = 0; r < 8; ++r)
                outb[(mt * 16 + laneH * 8 + r) * 132 + j] = acc[r];
        }
    }
    __syncthreads();

    // ---- Phase 6: masked, coalesced store ----
    #pragma unroll
    for (int i = 0; i < 32; ++i) {
        int e = i * 256 + t;
        int l = e >> 7, c = e & 127;
        out[(size_t)b * (LMAX * DM) + e] = (l < n) ? outb[l * 132 + c] : 0.0f;
    }
}

extern "C" void kernel_launch(void* const* d_in, const int* in_sizes, int n_in,
                              void* d_out, int out_size, void* d_ws, size_t ws_size,
                              hipStream_t stream) {
    const float* att_in   = (const float*)d_in[0];
    const float* Wqkv     = (const float*)d_in[1];   // [384][128]
    const float* bqkv     = (const float*)d_in[2];   // [384]
    const float* Wout     = (const float*)d_in[3];   // [128][128]
    const float* bout     = (const float*)d_in[4];   // [128]
    const int*   counts   = (const int*)d_in[5];     // [1024]
    float*       out      = (float*)d_out;

    // workspace: offsets | Wqkv f16 | Wout f16
    int*       offsets = (int*)d_ws;
    _Float16*  Wqkvh   = (_Float16*)((char*)d_ws + 4096);
    _Float16*  Wouth   = (_Float16*)((char*)d_ws + 4096 + 3 * DM * DM * sizeof(_Float16));

    (void)hipFuncSetAttribute(reinterpret_cast<const void*>(mhsa_kernel),
                              hipFuncAttributeMaxDynamicSharedMemorySize, SM_TOT);

    scan_offsets<<<1, NB, 0, stream>>>(counts, offsets);
    convert_weights<<<(4 * DM * DM + 255) / 256, 256, 0, stream>>>(Wqkv, Wout, Wqkvh, Wouth);
    mhsa_kernel<<<NB, 256, SM_TOT, stream>>>(att_in, bqkv, bout, counts, offsets,
                                             Wqkvh, Wouth, out);
}
